// FineMatching2_30099130810417
// MI455X (gfx1250) — matrix operand order repaired and checked by reading, not compiled, not run on hardware
//
#include <hip/hip_runtime.h>
#include <hip/hip_bf16.h>

typedef __attribute__((ext_vector_type(16))) _Float16 v16h;
typedef __attribute__((ext_vector_type(8)))  _Float16 v8h;
typedef __attribute__((ext_vector_type(4)))  _Float16 v4h;
typedef __attribute__((ext_vector_type(8)))  float    v8f;
typedef __attribute__((ext_vector_type(4)))  float    v4f;

#define NWIN   8192
#define WW     64
#define CC     128
#define LDS_K  (CC + 8)     // padded f16 row stride (272B -> rotating banks)
#define SIM_LD (WW + 1)     // padded f32 row stride for sim/conf
#define THR_V  0.1f

__global__ __launch_bounds__(128) void fine_matching_kernel(
    const float* __restrict__ f0g, const float* __restrict__ f1g,
    float* __restrict__ conf_out, float* __restrict__ mask_out,
    float* __restrict__ maskv_out, float* __restrict__ jids_out,
    float* __restrict__ mconf_out)
{
    __shared__ _Float16 ldsA[WW * LDS_K];
    __shared__ _Float16 ldsB[WW * LDS_K];
    __shared__ float    sim[WW * SIM_LD];     // sim, then conf in-place
    __shared__ float    rowM[WW], rowS[WW], colM[WW], colS[WW];
    __shared__ float    cRowMax[WW];
    __shared__ int      cRowJ[WW];
    __shared__ int      s_li, s_si, s_valid;
    __shared__ float    s_gm;

    const int tid = threadIdx.x;
    const int win = blockIdx.x;
    const float inv_sqrt_c = 0.08838834764831845f;   // 1/sqrt(128)
    const float invT       = 10.0f;                  // 1/TEMPERATURE

    // ---- Phase 1: global f32 -> scale -> f16 -> LDS (both matrices) ----
    const float* g0 = f0g + (size_t)win * WW * CC;
    const float* g1 = f1g + (size_t)win * WW * CC;
    for (int i = tid; i < (WW * CC) / 4; i += 128) {
        const int row = i >> 5;                // (i*4)/128
        const int col = (i & 31) << 2;         // (i*4)%128
        v4f a = *(const v4f*)(g0 + (size_t)i * 4);
        v4f b = *(const v4f*)(g1 + (size_t)i * 4);
        v4h ha, hb;
        #pragma unroll
        for (int j = 0; j < 4; ++j) {
            ha[j] = (_Float16)(a[j] * inv_sqrt_c);
            hb[j] = (_Float16)(b[j] * inv_sqrt_c);
        }
        *(v4h*)&ldsA[row * LDS_K + col] = ha;
        *(v4h*)&ldsB[row * LDS_K + col] = hb;
    }
    __syncthreads();

    // ---- Phase 2: per-wave 16x64 strip of sim via v_wmma_f32_16x16x32_f16 ----
    const int wave  = tid >> 5;
    const int lane  = tid & 31;
    const int lhalf = (lane < 16) ? 0 : 1;
    const int ln    = lane & 15;

    v8f acc[4];
    #pragma unroll
    for (int t = 0; t < 4; ++t) acc[t] = (v8f){0.f,0.f,0.f,0.f,0.f,0.f,0.f,0.f};

    const int arow = wave * 16 + ln;           // A-matrix row owned by this lane
    #pragma unroll
    for (int kk = 0; kk < 4; ++kk) {
        const int kbase = kk * 32;
        // A fragment (16-bit A layout: halves own K {0..7}/{8..15}, then +16)
        const int aoff = kbase + lhalf * 8;
        v8h a0 = *(const v8h*)&ldsA[arow * LDS_K + aoff];
        v8h a1 = *(const v8h*)&ldsA[arow * LDS_K + aoff + 16];
        v16h afrag = __builtin_shufflevector(a0, a1,
            0,1,2,3,4,5,6,7,8,9,10,11,12,13,14,15);

        const int boff = kbase + lhalf * 8;    // B halves own K {0..7}/{8..15}, then +16
        #pragma unroll
        for (int t = 0; t < 4; ++t) {
            const int brow = t * 16 + ln;      // B = f1^T: lane n holds f1 row n
            v8h b0 = *(const v8h*)&ldsB[brow * LDS_K + boff];
            v8h b1 = *(const v8h*)&ldsB[brow * LDS_K + boff + 16];
            v16h bfrag = __builtin_shufflevector(b0, b1,
                0,1,2,3,4,5,6,7,8,9,10,11,12,13,14,15);
            acc[t] = __builtin_amdgcn_wmma_f32_16x16x32_f16(
                false, afrag, false, bfrag, (short)0, acc[t], false, false);
        }
    }

    // ---- Phase 3: spill sim (scaled by 1/T) to LDS ----
    #pragma unroll
    for (int t = 0; t < 4; ++t) {
        #pragma unroll
        for (int r = 0; r < 8; ++r) {
            const int row = wave * 16 + lhalf * 8 + r;   // C/D layout
            const int col = t * 16 + ln;
            sim[row * SIM_LD + col] = acc[t][r] * invT;
        }
    }
    __syncthreads();

    // ---- Phase 4: softmax stats (rows by tid<64, cols by tid>=64) ----
    if (tid < 64) {
        const int l = tid;
        float m = -INFINITY;
        for (int s = 0; s < WW; ++s) m = fmaxf(m, sim[l * SIM_LD + s]);
        float sum = 0.f;
        for (int s = 0; s < WW; ++s) sum += __expf(sim[l * SIM_LD + s] - m);
        rowM[l] = m; rowS[l] = sum;
    } else {
        const int s = tid - 64;
        float m = -INFINITY;
        for (int l = 0; l < WW; ++l) m = fmaxf(m, sim[l * SIM_LD + s]);
        float sum = 0.f;
        for (int l = 0; l < WW; ++l) sum += __expf(sim[l * SIM_LD + s] - m);
        colM[s] = m; colS[s] = sum;
    }
    __syncthreads();

    // ---- Phase 5: conf = softmax1 * softmax2, stream to global, keep in LDS ----
    float* confW = conf_out + (size_t)win * WW * WW;
    for (int e = tid; e < WW * WW; e += 128) {
        const int l = e >> 6, s = e & 63;
        const float v = sim[l * SIM_LD + s];
        const float c = __expf(2.f * v - rowM[l] - colM[s]) / (rowS[l] * colS[s]);
        sim[l * SIM_LD + s] = c;
        confW[e] = c;
    }
    __syncthreads();

    // ---- Phase 6: global argmax with first-occurrence (row-major) tie-break ----
    if (tid < 64) {
        const int l = tid;
        float m = -1.0f; int j = 0;
        for (int s = 0; s < WW; ++s) {
            const float c = sim[l * SIM_LD + s];
            if (c > m) { m = c; j = s; }       // strict '>' => first occurrence
        }
        cRowMax[l] = m; cRowJ[l] = j;
    }
    __syncthreads();
    if (tid == 0) {
        float gm = -1.0f; int li = 0;
        for (int l = 0; l < WW; ++l)
            if (cRowMax[l] > gm) { gm = cRowMax[l]; li = l; }
        s_gm = gm; s_li = li; s_si = cRowJ[li];
        // global max entry automatically equals its row-max and col-max,
        // so mask reduces to (global max > THR) at the argmax location.
        s_valid = (gm > THR_V) ? 1 : 0;
    }
    __syncthreads();

    // ---- Phase 7: write mask / mask_v / j_ids / mconf (full regions, every call) ----
    float* maskW = mask_out + (size_t)win * WW * WW;
    const int flat = s_li * WW + s_si;
    const int valid = s_valid;
    for (int e = tid; e < WW * WW; e += 128)
        maskW[e] = (valid && e == flat) ? 1.0f : 0.0f;

    if (tid < 64) {
        const int l = tid;
        const int hit = (valid && l == s_li);
        maskv_out[(size_t)win * WW + l] = hit ? 1.0f : 0.0f;
        jids_out [(size_t)win * WW + l] = hit ? (float)s_si : 0.0f;
        mconf_out[(size_t)win * WW + l] = hit ? s_gm : 0.0f;
    }
}

extern "C" void kernel_launch(void* const* d_in, const int* in_sizes, int n_in,
                              void* d_out, int out_size, void* d_ws, size_t ws_size,
                              hipStream_t stream) {
    (void)in_sizes; (void)n_in; (void)out_size; (void)d_ws; (void)ws_size;
    const float* f0 = (const float*)d_in[0];
    const float* f1 = (const float*)d_in[1];
    float* out = (float*)d_out;

    const size_t nconf = (size_t)NWIN * WW * WW;   // 33,554,432
    const size_t nrow  = (size_t)NWIN * WW;        //    524,288
    float* conf  = out;
    float* mask  = out + nconf;
    float* maskv = out + 2 * nconf;
    float* jids  = out + 2 * nconf + nrow;
    float* mconf = out + 2 * nconf + 2 * nrow;

    fine_matching_kernel<<<NWIN, 128, 0, stream>>>(f0, f1, conf, mask, maskv, jids, mconf);
}
